// CSGNN_40759239639749
// MI455X (gfx1250) — matrix operand order repaired and checked by reading, not compiled, not run on hardware
//
#include <hip/hip_runtime.h>
#include <hip/hip_bf16.h>

// ---------------- problem constants (match reference) ----------------
#define N_NODES  50000
#define N_EDGES  600000
#define N_GRAPHS 256
#define F_NODE   92
#define F_EDGE   50
#define H        128
#define N_LAYERS 3

typedef __attribute__((ext_vector_type(16))) _Float16 v16h;
typedef __attribute__((ext_vector_type(8)))  float    v8f;

#define WMMA(a, b, c) \
  __builtin_amdgcn_wmma_f32_16x16x32_f16(false, (a), false, (b), (short)0, (c), false, false)

// =====================================================================
// Fragment helpers (layouts per ISA 7.12.2, wave32)
//   A 16x32 f16 : lanes 0-15: M=lane, K {0..7,16..23}; lanes 16-31: M=lane-16, K {8..15,24..31}
//   B 32x16 f16 : lanes 0-15: N=lane, K {0..7,16..23}; lanes 16-31: N=lane-16, K {8..15,24..31}
//   C 16x16 f32 : VGPR j: lanes 0-15 M=j, lanes 16-31 M=j+8, N=lane&15
// =====================================================================

// ---- A fragment from row-major f32 global, zero-pad K>=kmax
__device__ __forceinline__ v16h a_frag_f32(const float* __restrict__ A, int ld,
                                           int row0, int kbase, int kmax) {
  int lane = threadIdx.x & 31;
  const float* ap = A + (size_t)(row0 + (lane & 15)) * ld;
  int kh = (lane >> 4) << 3;  // 0 or 8
  v16h a;
#pragma unroll
  for (int i = 0; i < 8; ++i) {
    int k0 = kbase + kh + i;
    int k1 = k0 + 16;
    a[i]     = (k0 < kmax) ? (_Float16)ap[k0] : (_Float16)0.0f;
    a[i + 8] = (k1 < kmax) ? (_Float16)ap[k1] : (_Float16)0.0f;
  }
  return a;
}

// ---- per-element (guarded) B fragment from raw row-major f16 LDS weights
__device__ __forceinline__ v16h b_frag_raw(const _Float16* W, int kbase, int n0, int kmax) {
  int lane = threadIdx.x & 31;
  int n = n0 + (lane & 15);
  int koff = kbase + ((lane >> 4) << 3);
  v16h b;
#pragma unroll
  for (int i = 0; i < 16; ++i) {
    int k = koff + (i & 7) + ((i >> 3) << 4);
    b[i] = (k < kmax) ? W[k * H + n] : (_Float16)0.0f;
  }
  return b;
}

// ---- cooperative fill of LDS weight buffer PRE-SWIZZLED into B-fragment
// layout: Wf[kt][nt][lane][16] f16; consumer reads one contiguous v16h per frag
__device__ __forceinline__ void fill_wfrag(_Float16* __restrict__ Wf,
                                           const float* __restrict__ W,
                                           int ktiles, int kmax) {
  int total = ktiles << 12;  // ktiles*8*32*16
  for (int d = threadIdx.x; d < total; d += blockDim.x) {
    int i    = d & 15;
    int lane = (d >> 4) & 31;
    int nt   = (d >> 9) & 7;
    int kt   = d >> 12;
    int n = (nt << 4) + (lane & 15);
    int k = (kt << 5) + ((lane >> 4) << 3) + (i & 7) + ((i >> 3) << 4);
    Wf[d] = (k < kmax) ? (_Float16)W[k * H + n] : (_Float16)0.0f;
  }
}

__device__ __forceinline__ v16h b_frag(const _Float16* Wf, int kt, int nt) {
  int lane = threadIdx.x & 31;
  return *(const v16h*)(Wf + ((((kt << 3) + nt) << 5) + lane) * 16);
}

// ---- scatter one A-tile element (row m, col k) into swizzled A-frag LDS buffer
// Uf[kt][lane][16]; consumer: a = *(v16h*)(Uf + kt*512 + lane*16)
__device__ __forceinline__ int a_swizzle_idx(int m, int k) {
  int kt = k >> 5, kk = k & 31;
  int lp = (m & 15) + (((kk >> 3) & 1) << 4);
  int ip = (kk & 7) + ((kk >> 4) << 3);
  return (kt << 9) + (lp << 4) + ip;
}

// =====================================================================
// h = x @ W_node + b_node      [N_NODES x F_NODE] @ [F_NODE x H]
// =====================================================================
__global__ __launch_bounds__(256) void embed_nodes_kernel(
    const float* __restrict__ x, const float* __restrict__ Wn,
    const float* __restrict__ bn, float* __restrict__ h) {
  __shared__ __align__(32) _Float16 Wf[3 * 8 * 32 * 16];  // 24576 B, swizzled
  __shared__ float bl[H];
  fill_wfrag(Wf, Wn, 3, F_NODE);
  if (threadIdx.x < H) bl[threadIdx.x] = bn[threadIdx.x];
  __syncthreads();

  int wave = threadIdx.x >> 5;
  int row0 = (blockIdx.x * 8 + wave) * 16;
  if (row0 >= N_NODES) return;

  v16h a0 = a_frag_f32(x, F_NODE, row0, 0, F_NODE);
  v16h a1 = a_frag_f32(x, F_NODE, row0, 32, F_NODE);
  v16h a2 = a_frag_f32(x, F_NODE, row0, 64, F_NODE);

  int lane = threadIdx.x & 31;
  int nlo = lane & 15, moff = (lane >> 4) << 3;
#pragma unroll
  for (int nt = 0; nt < 8; ++nt) {
    v8f acc = {};
    acc = WMMA(a0, b_frag(Wf, 0, nt), acc);
    acc = WMMA(a1, b_frag(Wf, 1, nt), acc);
    acc = WMMA(a2, b_frag(Wf, 2, nt), acc);
    float bias = bl[nt * 16 + nlo];
#pragma unroll
    for (int j = 0; j < 8; ++j)
      h[(size_t)(row0 + moff + j) * H + nt * 16 + nlo] = acc[j] + bias;
  }
}

// =====================================================================
// e = relu(ea @ W1 + b1) @ W2 + b2   (fused 2-layer edge MLP, f16 out)
// 128 threads = 4 waves; hidden tile staged in LDS already in A-frag layout.
// LDS: 12800 (W1 raw) + 32768 (W2 swizzled) + 1024 + 16384 (U) = 62976 B
// =====================================================================
__global__ __launch_bounds__(128) void edge_mlp_kernel(
    const float* __restrict__ ea,
    const float* __restrict__ W1, const float* __restrict__ b1,
    const float* __restrict__ W2, const float* __restrict__ b2,
    _Float16* __restrict__ e_out) {
  __shared__ __align__(32) _Float16 W1l[F_EDGE * H];        // raw row-major
  __shared__ __align__(32) _Float16 W2f[4 * 8 * 32 * 16];   // swizzled B-frags
  __shared__ float b1l[H], b2l[H];
  __shared__ __align__(32) _Float16 Uf[4][4 * 32 * 16];     // swizzled A-frags / wave
  for (int i = threadIdx.x; i < F_EDGE * H; i += blockDim.x) W1l[i] = (_Float16)W1[i];
  fill_wfrag(W2f, W2, 4, H);
  if (threadIdx.x < H) { b1l[threadIdx.x] = b1[threadIdx.x]; b2l[threadIdx.x] = b2[threadIdx.x]; }
  __syncthreads();

  int wave = threadIdx.x >> 5;
  int row0 = (blockIdx.x * 4 + wave) * 16;   // grid = E/64 exactly
  int lane = threadIdx.x & 31;
  int nlo = lane & 15, moff = (lane >> 4) << 3;
  _Float16* U = &Uf[wave][0];

  // ---- layer 1: u = relu(ea @ W1 + b1), K=50 zero-padded
  v16h a0 = a_frag_f32(ea, F_EDGE, row0, 0, F_EDGE);
  v16h a1 = a_frag_f32(ea, F_EDGE, row0, 32, F_EDGE);
#pragma unroll
  for (int nt = 0; nt < 8; ++nt) {
    v8f acc = {};
    acc = WMMA(a0, b_frag_raw(W1l, 0, nt * 16, F_EDGE), acc);
    acc = WMMA(a1, b_frag_raw(W1l, 32, nt * 16, F_EDGE), acc);
    float bias = b1l[nt * 16 + nlo];
#pragma unroll
    for (int j = 0; j < 8; ++j) {
      float v = acc[j] + bias;
      U[a_swizzle_idx(moff + j, nt * 16 + nlo)] = (_Float16)(v > 0.0f ? v : 0.0f);
    }
  }
  __syncthreads();  // C-layout scatter -> A-frag layout visible

  // ---- layer 2: e = u @ W2 + b2, K=128; A frags are contiguous v16h now
  v16h c0 = *(const v16h*)(U + 0 * 512 + lane * 16);
  v16h c1 = *(const v16h*)(U + 1 * 512 + lane * 16);
  v16h c2 = *(const v16h*)(U + 2 * 512 + lane * 16);
  v16h c3 = *(const v16h*)(U + 3 * 512 + lane * 16);
#pragma unroll
  for (int nt = 0; nt < 8; ++nt) {
    v8f acc = {};
    acc = WMMA(c0, b_frag(W2f, 0, nt), acc);
    acc = WMMA(c1, b_frag(W2f, 1, nt), acc);
    acc = WMMA(c2, b_frag(W2f, 2, nt), acc);
    acc = WMMA(c3, b_frag(W2f, 3, nt), acc);
    float bias = b2l[nt * 16 + nlo];
#pragma unroll
    for (int j = 0; j < 8; ++j)
      __builtin_nontemporal_store(
          (_Float16)(acc[j] + bias),
          e_out + (size_t)(row0 + moff + j) * H + nt * 16 + nlo);
  }
}

// =====================================================================
// t = h @ W        [N_NODES x H] @ [H x H], no bias
// =====================================================================
__global__ __launch_bounds__(256) void transform_kernel(
    const float* __restrict__ hin, const float* __restrict__ W,
    float* __restrict__ tout) {
  __shared__ __align__(32) _Float16 Wf[4 * 8 * 32 * 16];  // 32768 B swizzled
  fill_wfrag(Wf, W, 4, H);
  __syncthreads();

  int wave = threadIdx.x >> 5;
  int row0 = (blockIdx.x * 8 + wave) * 16;
  if (row0 >= N_NODES) return;

  v16h a0 = a_frag_f32(hin, H, row0, 0, H);
  v16h a1 = a_frag_f32(hin, H, row0, 32, H);
  v16h a2 = a_frag_f32(hin, H, row0, 64, H);
  v16h a3 = a_frag_f32(hin, H, row0, 96, H);

  int lane = threadIdx.x & 31;
  int nlo = lane & 15, moff = (lane >> 4) << 3;
#pragma unroll
  for (int nt = 0; nt < 8; ++nt) {
    v8f acc = {};
    acc = WMMA(a0, b_frag(Wf, 0, nt), acc);
    acc = WMMA(a1, b_frag(Wf, 1, nt), acc);
    acc = WMMA(a2, b_frag(Wf, 2, nt), acc);
    acc = WMMA(a3, b_frag(Wf, 3, nt), acc);
#pragma unroll
    for (int j = 0; j < 8; ++j)
      tout[(size_t)(row0 + moff + j) * H + nt * 16 + nlo] = acc[j];
  }
}

// ---------------- elementwise / scatter kernels ----------------------
__global__ void zero_kernel(float* __restrict__ p, int n) {
  int i = blockIdx.x * blockDim.x + threadIdx.x;
  if (i < n) p[i] = 0.0f;
}

__global__ void deg_count_kernel(const int* __restrict__ dst, float* __restrict__ deg) {
  int e = blockIdx.x * blockDim.x + threadIdx.x;
  if (e < N_EDGES) atomicAdd(&deg[dst[e]], 1.0f);
}

__global__ void dis_kernel(float* __restrict__ deg_to_dis) {
  int n = blockIdx.x * blockDim.x + threadIdx.x;
  if (n < N_NODES) deg_to_dis[n] = rsqrtf(deg_to_dis[n] + 1.0f);  // (deg+1)^-1/2
}

// agg[dst] += t[src] * e * (dis[src]*dis[dst]); one thread = 4 feats of one edge
__global__ void aggregate_kernel(const int* __restrict__ src, const int* __restrict__ dst,
                                 const _Float16* __restrict__ e, const float* __restrict__ t,
                                 const float* __restrict__ dis, float* __restrict__ agg) {
  int idx = blockIdx.x * blockDim.x + threadIdx.x;
  int eid = idx >> 5;
  if (eid >= N_EDGES) return;
  int f4 = (idx & 31) * 4;
  int s = src[eid], d = dst[eid];
  float nrm = dis[s] * dis[d];
  const float* tp = t + (size_t)s * H + f4;
  const _Float16* ep = e + (size_t)eid * H + f4;
  float* ap = agg + (size_t)d * H + f4;
#pragma unroll
  for (int i = 0; i < 4; ++i) {
    float ev = (float)__builtin_nontemporal_load(ep + i);  // stream e, keep t/agg in L2
    atomicAdd(ap + i, tp[i] * ev * nrm);
  }
}

// h = relu(agg + t*dis^2 + b_conv)
__global__ void selfloop_relu_kernel(const float* __restrict__ agg, const float* __restrict__ t,
                                     const float* __restrict__ dis, const float* __restrict__ bconv,
                                     float* __restrict__ hout) {
  int idx = blockIdx.x * blockDim.x + threadIdx.x;
  if (idx >= N_NODES * H) return;
  int n = idx >> 7, f = idx & (H - 1);
  float di = dis[n];
  float v = agg[idx] + t[idx] * di * di + bconv[f];
  hout[idx] = v > 0.0f ? v : 0.0f;
}

__global__ void pool_sum_kernel(const float* __restrict__ h, const int* __restrict__ batch,
                                float* __restrict__ g, float* __restrict__ counts) {
  int idx = blockIdx.x * blockDim.x + threadIdx.x;
  if (idx >= N_NODES * H) return;
  int n = idx >> 7, f = idx & (H - 1);
  int b = batch[n];
  atomicAdd(&g[(size_t)b * H + f], h[idx]);
  if (f == 0) atomicAdd(&counts[b], 1.0f);
}

__global__ void pool_div_kernel(float* __restrict__ g, const float* __restrict__ counts) {
  int idx = blockIdx.x * blockDim.x + threadIdx.x;
  if (idx >= N_GRAPHS * H) return;
  g[idx] /= fmaxf(counts[idx >> 7], 1.0f);
}

// out = relu(g@W_l1 + b_l1) @ W_l2 + b_l2 ; one block (128 thr) per graph
__global__ __launch_bounds__(128) void head_kernel(
    const float* __restrict__ g,
    const float* __restrict__ W1, const float* __restrict__ b1,
    const float* __restrict__ W2, const float* __restrict__ b2,
    float* __restrict__ out) {
  __shared__ float gr[H];
  __shared__ float red[H];
  int b = blockIdx.x, f = threadIdx.x;
  gr[f] = g[(size_t)b * H + f];
  __syncthreads();
  float s = b1[f];
#pragma unroll 8
  for (int k = 0; k < H; ++k) s += gr[k] * W1[k * H + f];
  red[f] = fmaxf(s, 0.0f) * W2[f];
  __syncthreads();
  for (int st = H / 2; st > 0; st >>= 1) {
    if (f < st) red[f] += red[f + st];
    __syncthreads();
  }
  if (f == 0) out[b] = red[0] + b2[0];
}

// =====================================================================
extern "C" void kernel_launch(void* const* d_in, const int* in_sizes, int n_in,
                              void* d_out, int out_size, void* d_ws, size_t ws_size,
                              hipStream_t stream) {
  const float* x         = (const float*)d_in[0];
  const float* edge_attr = (const float*)d_in[1];
  const int*   edge_idx  = (const int*)d_in[2];   // [2][E]
  const int*   batch     = (const int*)d_in[3];
  const float* W_node    = (const float*)d_in[4];
  const float* b_node    = (const float*)d_in[5];
  const float* W_e1      = (const float*)d_in[6];
  const float* b_e1      = (const float*)d_in[7];
  const float* W_e2      = (const float*)d_in[8];
  const float* b_e2      = (const float*)d_in[9];
  const float* W_convs   = (const float*)d_in[10]; // [3][H][H]
  const float* b_convs   = (const float*)d_in[11]; // [3][H]
  const float* W_l1      = (const float*)d_in[12];
  const float* b_l1      = (const float*)d_in[13];
  const float* W_l2      = (const float*)d_in[14];
  const float* b_l2      = (const float*)d_in[15];
  float* out = (float*)d_out;

  const int* src = edge_idx;
  const int* dst = edge_idx + N_EDGES;

  // -------- workspace layout (~231 MB total) --------
  char* ws = (char*)d_ws;
  float*     h    = (float*)ws;    ws += (size_t)N_NODES * H * 4;   // 25.6 MB
  float*     t    = (float*)ws;    ws += (size_t)N_NODES * H * 4;   // 25.6 MB
  float*     agg  = (float*)ws;    ws += (size_t)N_NODES * H * 4;   // 25.6 MB
  _Float16*  e    = (_Float16*)ws; ws += (size_t)N_EDGES * H * 2;   // 153.6 MB
  float*     dis  = (float*)ws;    ws += ((size_t)N_NODES * 4 + 255) & ~(size_t)255;
  float*     g    = (float*)ws;    ws += (size_t)N_GRAPHS * H * 4;
  float*     cnts = (float*)ws;    ws += (size_t)N_GRAPHS * 4;
  (void)ws_size; (void)n_in; (void)in_sizes; (void)out_size;

  const int TPB = 256;

  // degrees -> dis = (deg+1)^(-1/2)
  zero_kernel<<<(N_NODES + TPB - 1) / TPB, TPB, 0, stream>>>(dis, N_NODES);
  deg_count_kernel<<<(N_EDGES + TPB - 1) / TPB, TPB, 0, stream>>>(dst, dis);
  dis_kernel<<<(N_NODES + TPB - 1) / TPB, TPB, 0, stream>>>(dis);

  // embeddings
  embed_nodes_kernel<<<(N_NODES + 127) / 128, 256, 0, stream>>>(x, W_node, b_node, h);
  edge_mlp_kernel<<<N_EDGES / 64, 128, 0, stream>>>(edge_attr, W_e1, b_e1, W_e2, b_e2, e);

  // GCN layers
  for (int l = 0; l < N_LAYERS; ++l) {
    transform_kernel<<<(N_NODES + 127) / 128, 256, 0, stream>>>(h, W_convs + (size_t)l * H * H, t);
    zero_kernel<<<(N_NODES * H + TPB - 1) / TPB, TPB, 0, stream>>>(agg, N_NODES * H);
    aggregate_kernel<<<(N_EDGES * 32 + TPB - 1) / TPB, TPB, 0, stream>>>(src, dst, e, t, dis, agg);
    selfloop_relu_kernel<<<(N_NODES * H + TPB - 1) / TPB, TPB, 0, stream>>>(
        agg, t, dis, b_convs + (size_t)l * H, h);
  }

  // global mean pool + head
  zero_kernel<<<(N_GRAPHS * H + TPB - 1) / TPB, TPB, 0, stream>>>(g, N_GRAPHS * H);
  zero_kernel<<<1, TPB, 0, stream>>>(cnts, N_GRAPHS);
  pool_sum_kernel<<<(N_NODES * H + TPB - 1) / TPB, TPB, 0, stream>>>(h, batch, g, cnts);
  pool_div_kernel<<<(N_GRAPHS * H + TPB - 1) / TPB, TPB, 0, stream>>>(g, cnts);
  head_kernel<<<N_GRAPHS, H, 0, stream>>>(g, W_l1, b_l1, W_l2, b_l2, out);
}
